// MHSA_8538394985033
// MI455X (gfx1250) — matrix operand order repaired and checked by reading, not compiled, not run on hardware
//
#include <hip/hip_runtime.h>
#include <hip/hip_bf16.h>
#include <stdint.h>

#define B_  4
#define T_  1024
#define CM  256
#define CZ  128
#define H_  8
#define DH  32

typedef __attribute__((ext_vector_type(16))) __bf16        v16bf;
typedef __attribute__((ext_vector_type(8)))  float         v8f;
typedef __attribute__((ext_vector_type(4)))  float         f32x4;
typedef __attribute__((ext_vector_type(4)))  unsigned int  u32x4;
typedef __attribute__((ext_vector_type(8)))  unsigned int  u32x8;

// ---- fragment load helpers (all 16B-aligned vector loads) ----
static __device__ inline v16bf load_2x8(const __bf16* p0, const __bf16* p1) {
  u32x4 a = *(const u32x4*)p0;
  u32x4 b = *(const u32x4*)p1;
  u32x8 r = __builtin_shufflevector(a, b, 0, 1, 2, 3, 4, 5, 6, 7);
  return __builtin_bit_cast(v16bf, r);
}
static __device__ inline v16bf load_16(const __bf16* p) { return load_2x8(p, p + 8); }

static __device__ inline v8f wmma_bf16(v16bf a, v16bf b, v8f c) {
  return __builtin_amdgcn_wmma_f32_16x16x32_bf16(false, a, false, b, (short)0, c,
                                                 false, false);
}

// ---- prep: fp32 -> bf16 copy ----
__global__ void k_cvt(const float* __restrict__ in, __bf16* __restrict__ out, int n) {
  int i = blockIdx.x * blockDim.x + threadIdx.x;
  if (i < n) out[i] = (__bf16)in[i];
}

// ---- prep: transpose [K,N] fp32 -> [N,K] bf16 ----
__global__ void k_tr(const float* __restrict__ in, __bf16* __restrict__ out,
                     int K, int N) {
  int i = blockIdx.x * blockDim.x + threadIdx.x;
  if (i < K * N) {
    int k = i / N, n = i % N;
    out[(size_t)n * K + k] = (__bf16)in[i];
  }
}

// ---- pair-bias projection: bias[h][i][j] = bias_rep[i,j,:] @ Wb + bb  (streams 537MB) ----
__global__ void k_bias(const float* __restrict__ br, const float* __restrict__ Wb,
                       const float* __restrict__ bb, float* __restrict__ bias) {
  __shared__ float sW[CZ * H_];
  __shared__ float sb[H_];
  for (int i = threadIdx.x; i < CZ * H_; i += blockDim.x) sW[i] = Wb[i];
  if (threadIdx.x < H_) sb[threadIdx.x] = bb[threadIdx.x];
  __syncthreads();
  size_t idx = (size_t)blockIdx.x * blockDim.x + threadIdx.x;  // (i,j) flat < 2^20
  const float* rp = br + idx * CZ;
  f32x4 acc0 = {sb[0], sb[1], sb[2], sb[3]};
  f32x4 acc1 = {sb[4], sb[5], sb[6], sb[7]};
  for (int c = 0; c < CZ; c += 4) {
    f32x4 v = *(const f32x4*)(rp + c);
#pragma unroll
    for (int u = 0; u < 4; ++u) {
      f32x4 w0 = *(const f32x4*)(&sW[(c + u) * H_]);
      f32x4 w1 = *(const f32x4*)(&sW[(c + u) * H_ + 4]);
      acc0 += v[u] * w0;
      acc1 += v[u] * w1;
    }
  }
#pragma unroll
  for (int h = 0; h < 4; ++h) bias[((size_t)h << 20) + idx] = acc0[h];
#pragma unroll
  for (int h = 0; h < 4; ++h) bias[((size_t)(h + 4) << 20) + idx] = acc1[h];
}

// ---- QKV projection: [B*T,256] @ [256,1024], scatter to Q,K (scaled), Vt, sigmoid(G) ----
__global__ void k_qkv(const __bf16* __restrict__ xbf, const __bf16* __restrict__ wt,
                      __bf16* __restrict__ Q, __bf16* __restrict__ Km,
                      __bf16* __restrict__ Vt, float* __restrict__ Gs) {
  const int lane = threadIdx.x & 31;
  const int wav  = threadIdx.x >> 5;
  const int wid  = blockIdx.x * 4 + wav;   // 16384 waves
  const int mt = wid >> 6;                 // 0..255 (row tile over B*T)
  const int nt = wid & 63;                 // 0..63  (col tile over 1024)
  const int hi = lane >> 4, l16 = lane & 15;
  const __bf16* arow = xbf + (size_t)(mt * 16 + l16) * CM;
  const __bf16* brow = wt  + (size_t)(nt * 16 + l16) * CM;
  v8f c = {};
#pragma unroll
  for (int k0 = 0; k0 < CM; k0 += 32) {
    v16bf a = load_2x8(arow + k0 + 8 * hi, arow + k0 + 8 * hi + 16);
    v16bf b = load_2x8(brow + k0 + 8 * hi, brow + k0 + 8 * hi + 16);
    c = wmma_bf16(a, b, c);
  }
  // channel decode: cc = d*32 + kk*8 + h  (reference reshape (d,4,H))
  const int cc = nt * 16 + l16;
  const int d = cc >> 5, kk = (cc >> 3) & 3, h = cc & 7;
#pragma unroll
  for (int r = 0; r < 8; ++r) {
    int grow = mt * 16 + r + 8 * hi;
    int b_ = grow >> 10, t = grow & (T_ - 1);
    int bh = b_ * H_ + h;
    float v = c[r];
    if (kk == 0)      Q [((size_t)bh * T_ + t) * DH + d] = (__bf16)(v * 0.17677669529663689f);
    else if (kk == 1) Km[((size_t)bh * T_ + t) * DH + d] = (__bf16)v;
    else if (kk == 2) Vt[((size_t)bh * DH + d) * T_ + t] = (__bf16)v;
    else              Gs[((size_t)bh * T_ + t) * DH + d] = 1.0f / (1.0f + __expf(-v));
  }
}

// ---- flash attention in S^T orientation; one wave per 16-query tile ----
__global__ void k_attn(const __bf16* __restrict__ Q, const __bf16* __restrict__ Km,
                       const __bf16* __restrict__ Vt, const float* __restrict__ Gs,
                       const float* __restrict__ bias, __bf16* __restrict__ AO) {
  const int lane = threadIdx.x & 31;
  const int wav = threadIdx.x >> 5;
  const int wid = blockIdx.x * 4 + wav;    // 2048 waves
  const int qt = wid & 63;                 // query tile
  const int bh = wid >> 6;
  const int b = bh >> 3, h = bh & 7;
  const int hi = lane >> 4, l16 = lane & 15;
  const int iq = qt * 16 + l16;            // this lane's query (column index of S^T)

  // B-fragment of Q^T: lane-lo holds d0..7,16..23, lane-hi d8..15,24..31 of its query row
  v16bf bQ = load_2x8(Q + ((size_t)bh * T_ + iq) * DH + 8 * hi, Q + ((size_t)bh * T_ + iq) * DH + 8 * hi + 16);
  v8f o0 = {}, o1 = {};
  float m = -3.0e38f, l = 0.0f;
  const float*  brow  = bias + ((size_t)(h * T_ + iq)) * T_;
  const __bf16* kbase = Km + (size_t)bh * T_ * DH;
  const __bf16* vbase = Vt + (size_t)bh * DH * T_ + (size_t)l16 * T_;

  for (int j0 = 0; j0 < T_; j0 += 32) {
    // A-fragments of K rows (two 16-key blocks)
    const __bf16* kr0 = kbase + (size_t)(j0 + l16) * DH + 8 * hi;
    const __bf16* kr1 = kr0 + 16 * DH;
    v16bf aK0 = load_2x8(kr0, kr0 + 16);
    v16bf aK1 = load_2x8(kr1, kr1 + 16);
    v8f z = {};
    v8f s0 = wmma_bf16(aK0, bQ, z);  // S^T: key = j0 + r + 8*hi, query = l16
    v8f s1 = wmma_bf16(aK1, bQ, z);  // keys j0+16..31

    // bias rows: contiguous 8 keys per lane-half
    f32x4 b00 = *(const f32x4*)(brow + j0 + 8 * hi);
    f32x4 b01 = *(const f32x4*)(brow + j0 + 8 * hi + 4);
    f32x4 b10 = *(const f32x4*)(brow + j0 + 16 + 8 * hi);
    f32x4 b11 = *(const f32x4*)(brow + j0 + 16 + 8 * hi + 4);

    float x0[8], x1[8], mn = m;
#pragma unroll
    for (int r = 0; r < 8; ++r) {
      x0[r] = s0[r] + ((r < 4) ? b00[r] : b01[r - 4]);
      x1[r] = s1[r] + ((r < 4) ? b10[r] : b11[r - 4]);
      mn = fmaxf(mn, fmaxf(x0[r], x1[r]));
    }
    mn = fmaxf(mn, __shfl_xor(mn, 16, 32));  // combine key-halves (same query)
    float corr = __expf(m - mn);
    m = mn;
    float p0[8], p1[8], ls = 0.0f;
#pragma unroll
    for (int r = 0; r < 8; ++r) {
      p0[r] = __expf(x0[r] - mn);
      p1[r] = __expf(x1[r] - mn);
      ls += p0[r] + p1[r];
    }
    ls += __shfl_xor(ls, 16, 32);
    l = l * corr + ls;
#pragma unroll
    for (int r = 0; r < 8; ++r) { o0[r] *= corr; o1[r] *= corr; }

    // Build P^T B-fragment: half-wave swap moves keys 8..15 / 16..23 into place
    v16bf bP;
#pragma unroll
    for (int r = 0; r < 8; ++r) {
      float t0x = __shfl_xor(p0[r], 16, 32);
      float t1x = __shfl_xor(p1[r], 16, 32);
      float alo = hi ? t1x : p0[r];   // lo: key r      | hi: key 16+r
      float ahi = hi ? p1[r] : t0x;   // lo: key 8+r    | hi: key 24+r
      bP[r]     = (__bf16)alo;
      bP[8 + r] = (__bf16)ahi;
    }

    // A-fragments of V^T (d rows, 32 keys)
    const __bf16* v0p = vbase + j0 + 16 * hi;
    const __bf16* v1p = v0p + 16 * T_;
    v16bf aV0 = load_16(v0p);
    v16bf aV1 = load_16(v1p);
    o0 = wmma_bf16(aV0, bP, o0);  // O^T[d0..15,  q]
    o1 = wmma_bf16(aV1, bP, o1);  // O^T[d16..31, q]
  }

  float invl = 1.0f / l;
  const float* gp = Gs + ((size_t)bh * T_ + iq) * DH + 8 * hi;
  f32x4 g0 = *(const f32x4*)gp;
  f32x4 g1 = *(const f32x4*)(gp + 4);
  f32x4 g2 = *(const f32x4*)(gp + 16);
  f32x4 g3 = *(const f32x4*)(gp + 20);
  __bf16* ob = AO + ((size_t)(b * T_ + iq)) * (H_ * DH) + h * DH;
  __bf16 t0[8] __attribute__((aligned(16)));
  __bf16 t1[8] __attribute__((aligned(16)));
#pragma unroll
  for (int r = 0; r < 8; ++r) {
    float gv0 = (r < 4) ? g0[r] : g1[r - 4];
    float gv1 = (r < 4) ? g2[r] : g3[r - 4];
    t0[r] = (__bf16)(o0[r] * invl * gv0);
    t1[r] = (__bf16)(o1[r] * invl * gv1);
  }
  *(u32x4*)(ob + 8 * hi)      = *(const u32x4*)t0;
  *(u32x4*)(ob + 16 + 8 * hi) = *(const u32x4*)t1;
}

// ---- output projection: [B*T,256] @ W0 ----
__global__ void k_out(const __bf16* __restrict__ AO, const __bf16* __restrict__ W0t,
                      float* __restrict__ out) {
  const int lane = threadIdx.x & 31;
  const int wav = threadIdx.x >> 5;
  const int wid = blockIdx.x * 4 + wav;    // 4096 waves
  const int mt = wid >> 4, nt = wid & 15;
  const int hi = lane >> 4, l16 = lane & 15;
  const __bf16* arow = AO  + (size_t)(mt * 16 + l16) * CM;
  const __bf16* brow = W0t + (size_t)(nt * 16 + l16) * CM;
  v8f c = {};
#pragma unroll
  for (int k0 = 0; k0 < CM; k0 += 32) {
    v16bf a = load_2x8(arow + k0 + 8 * hi, arow + k0 + 8 * hi + 16);
    v16bf b = load_2x8(brow + k0 + 8 * hi, brow + k0 + 8 * hi + 16);
    c = wmma_bf16(a, b, c);
  }
  const int n = nt * 16 + l16;
#pragma unroll
  for (int r = 0; r < 8; ++r)
    out[(size_t)(mt * 16 + r + 8 * hi) * CM + n] = c[r];
}

extern "C" void kernel_launch(void* const* d_in, const int* in_sizes, int n_in,
                              void* d_out, int out_size, void* d_ws, size_t ws_size,
                              hipStream_t stream) {
  const float* x        = (const float*)d_in[0];
  const float* bias_rep = (const float*)d_in[1];
  const float* Wqkv     = (const float*)d_in[2];
  const float* W0       = (const float*)d_in[3];
  const float* Wb       = (const float*)d_in[4];
  const float* bb       = (const float*)d_in[5];
  float* out = (float*)d_out;

  char* ws = (char*)d_ws;
  __bf16* xbf  = (__bf16*)(ws + 0);          //  2 MiB
  __bf16* wqt  = (__bf16*)(ws + 2097152);    //  512 KiB (Wqkv^T bf16)
  __bf16* w0t  = (__bf16*)(ws + 2621440);    //  128 KiB (W0^T bf16)
  __bf16* Q    = (__bf16*)(ws + 2752512);    //  2 MiB [B,H,T,DH]
  __bf16* Km   = (__bf16*)(ws + 4849664);    //  2 MiB [B,H,T,DH]
  __bf16* Vt   = (__bf16*)(ws + 6946816);    //  2 MiB [B,H,DH,T]
  float*  Gs   = (float* )(ws + 9043968);    //  4 MiB [B,H,T,DH] fp32 sigmoid(g)
  float*  bias = (float* )(ws + 13238272);   // 32 MiB [H,T,T] fp32
  __bf16* AO   = (__bf16*)(ws + 46792704);   //  2 MiB [B,T,H*DH]
  // total = 48,889,856 bytes

  k_cvt <<<4096, 256, 0, stream>>>(x, xbf, B_ * T_ * CM);
  k_tr  <<<1024, 256, 0, stream>>>(Wqkv, wqt, CM, 4 * H_ * DH);
  k_tr  <<<256,  256, 0, stream>>>(W0, w0t, CM, CM);
  k_bias<<<4096, 256, 0, stream>>>(bias_rep, Wb, bb, bias);
  k_qkv <<<4096, 128, 0, stream>>>(xbf, wqt, Q, Km, Vt, Gs);
  k_attn<<<512,  128, 0, stream>>>(Q, Km, Vt, Gs, bias, AO);
  k_out <<<1024, 128, 0, stream>>>(AO, w0t, out);
}
